// MSA_5669356830773
// MI455X (gfx1250) — matrix operand order repaired and checked by reading, not compiled, not run on hardware
//
#include <hip/hip_runtime.h>

// ---------------------------------------------------------------------------
// Fused Pre-LN MSA block for MI455X (gfx1250, wave32, WMMA 16x16x32 bf16)
//   z:[8,1024,1024] f32 -> LN -> QKV GEMM -> flash attention -> proj+residual
// All LDS layouts chosen so every WMMA fragment is two ds_load_b128 per lane.
// Tile staging uses global_load_async_to_lds_b128 (ASYNCcnt path).
// ---------------------------------------------------------------------------

typedef __attribute__((ext_vector_type(16))) __bf16 v16bf;
typedef __attribute__((ext_vector_type(8)))  __bf16 v8bf;
typedef __attribute__((ext_vector_type(8)))  float  v8f;

#define DEVINL static __device__ __forceinline__
#define USE_ASYNC_LDS 1

DEVINL __bf16 f2bf(float f) {
  unsigned u = __builtin_bit_cast(unsigned, f);
  u += 0x7FFFu + ((u >> 16) & 1u);            // round-to-nearest-even
  unsigned short s = (unsigned short)(u >> 16);
  return __builtin_bit_cast(__bf16, s);
}

DEVINL v8f v8f_zero() {
  v8f r;
#pragma unroll
  for (int i = 0; i < 8; ++i) r[i] = 0.0f;
  return r;
}

DEVINL v16bf concat8(v8bf lo, v8bf hi) {
  return __builtin_shufflevector(lo, hi, 0, 1, 2, 3, 4, 5, 6, 7,
                                         8, 9, 10, 11, 12, 13, 14, 15);
}

// A-fragment (16x32 bf16, ISA 7.12.2): lane group g holds K runs [8g,8g+7] and
// [16+8g, 23+8g] for its row -> two contiguous 16B LDS loads.
DEVINL v16bf ldsA_frag(const __bf16* row, int g) {
  v8bf lo = *(const v8bf*)(row + 8 * g);
  v8bf hi = *(const v8bf*)(row + 16 + 8 * g);
  return concat8(lo, hi);
}

// B-fragment (32x16 bf16): lane group g holds K runs [8g,8g+7] and
// [16+8g,23+8g] for its column (matches A) -> two 16B LDS loads.
DEVINL v16bf ldsB_frag(const __bf16* row, int g) {
  v8bf lo = *(const v8bf*)(row + 8 * g);
  v8bf hi = *(const v8bf*)(row + 16 + 8 * g);
  return concat8(lo, hi);
}

DEVINL v8f wmma_bf16(v16bf a, v16bf b, v8f c) {
  return __builtin_amdgcn_wmma_f32_16x16x32_bf16(
      false, a, false, b, (short)0, c, false, false);
}

// Per-lane async 16B global -> LDS copy (GLOBAL_LOAD_ASYNC_TO_LDS_B128,
// tracked by ASYNCcnt).  LDS operand = low 32 bits of the generic address
// (aperture scheme: addr[31:0] is the wave-relative LDS byte address).
DEVINL void cp16(void* sptr, const void* gptr) {
#if USE_ASYNC_LDS
  unsigned lds = (unsigned)(size_t)sptr;
  unsigned long long ga = (unsigned long long)(size_t)gptr;
  asm volatile("global_load_async_to_lds_b128 %0, %1, off"
               :: "v"(lds), "v"(ga) : "memory");
#else
  *(uint4*)sptr = *(const uint4*)gptr;
#endif
}
DEVINL void cp16_wait() {
#if USE_ASYNC_LDS
  asm volatile("s_wait_asynccnt 0" ::: "memory");
#endif
}

// ---------------------------------------------------------------------------
// 1) f32 -> bf16 convert + transpose (weights): dst[n][k] = bf16(src[k][n])
// ---------------------------------------------------------------------------
__global__ __launch_bounds__(256) void cvt_t(const float* __restrict__ src,
                                             __bf16* __restrict__ dst,
                                             int K, int N) {
  __shared__ __bf16 tile[32][33];
  const int n0 = blockIdx.x * 32, k0 = blockIdx.y * 32;
  const int tx = threadIdx.x & 31, ty = threadIdx.x >> 5;  // 32 x 8
#pragma unroll
  for (int i = 0; i < 4; ++i) {
    int k = ty + i * 8;
    tile[k][tx] = f2bf(src[(size_t)(k0 + k) * N + n0 + tx]);
  }
  __syncthreads();
#pragma unroll
  for (int i = 0; i < 4; ++i) {
    int n = ty + i * 8;
    dst[(size_t)(n0 + n) * K + k0 + tx] = tile[tx][n];
  }
}

// ---------------------------------------------------------------------------
// 2) LayerNorm: one block (256 threads) per token row of 1024
// ---------------------------------------------------------------------------
__global__ __launch_bounds__(256) void ln_kernel(const float* __restrict__ z,
                                                 const float* __restrict__ w,
                                                 const float* __restrict__ b,
                                                 __bf16* __restrict__ zn) {
  const int row = blockIdx.x, t = threadIdx.x;
  const float* zr = z + (size_t)row * 1024;
  float4 v = *(const float4*)(zr + t * 4);
  float s  = v.x + v.y + v.z + v.w;
  float ss = v.x * v.x + v.y * v.y + v.z * v.z + v.w * v.w;
  __shared__ float r1[256], r2[256];
  r1[t] = s; r2[t] = ss;
  __syncthreads();
  for (int o = 128; o > 0; o >>= 1) {
    if (t < o) { r1[t] += r1[t + o]; r2[t] += r2[t + o]; }
    __syncthreads();
  }
  float mu   = r1[0] * (1.0f / 1024.0f);
  float var  = r2[0] * (1.0f / 1024.0f) - mu * mu;
  float rstd = rsqrtf(var + 1e-5f);
  int c = t * 4;
  __bf16* o4 = zn + (size_t)row * 1024 + c;
  o4[0] = f2bf((v.x - mu) * rstd * w[c + 0] + b[c + 0]);
  o4[1] = f2bf((v.y - mu) * rstd * w[c + 1] + b[c + 1]);
  o4[2] = f2bf((v.z - mu) * rstd * w[c + 2] + b[c + 2]);
  o4[3] = f2bf((v.w - mu) * rstd * w[c + 3] + b[c + 3]);
}

// ---------------------------------------------------------------------------
// 3) QKV GEMM: C[8192,3072] = zn(bf16) x Wqkv^T(bf16, n-major) + b_qkv
//    Block tile 128x128, BK=32, 8 waves (4x2), 32x64 per wave (8 WMMA/K-step).
//    Epilogue scatters into qkv planes: q,k as [n][dh]; v transposed [dh][n].
// ---------------------------------------------------------------------------
__global__ __launch_bounds__(256) void qkv_gemm(const __bf16* __restrict__ A,
                                                const __bf16* __restrict__ Bt,
                                                const float* __restrict__ bias,
                                                __bf16* __restrict__ qkv) {
  __shared__ __bf16 sA[128][32];    // rows m, k-contig
  __shared__ __bf16 sB[128][32];    // rows n, k-contig (weights pre-transposed)
  const int t = threadIdx.x;
  const int lane = t & 31, wave = t >> 5;
  const int g = lane >> 4, ln = lane & 15;
  const int wm = wave >> 1, wn = wave & 1;
  const int m0 = blockIdx.x * 128, n0 = blockIdx.y * 128;

  v8f acc[2][4];
#pragma unroll
  for (int i = 0; i < 2; ++i)
#pragma unroll
    for (int j = 0; j < 4; ++j) acc[i][j] = v8f_zero();

  for (int kt = 0; kt < 1024; kt += 32) {
    __syncthreads();
#pragma unroll
    for (int it = 0; it < 2; ++it) {             // 512 chunks each tile
      int c = t + it * 256;
      int row = c >> 2, col = (c & 3) * 8;
      cp16(&sA[row][col], A  + (size_t)(m0 + row) * 1024 + kt + col);
      cp16(&sB[row][col], Bt + (size_t)(n0 + row) * 1024 + kt + col);
    }
    if (kt + 32 < 1024) {                        // prefetch next K-tile
      __builtin_prefetch(A  + (size_t)(m0 + (t >> 1)) * 1024 + kt + 32, 0, 3);
      __builtin_prefetch(Bt + (size_t)(n0 + (t >> 1)) * 1024 + kt + 32, 0, 3);
    }
    cp16_wait();
    __syncthreads();

    v16bf af[2], bf_[4];
#pragma unroll
    for (int fm = 0; fm < 2; ++fm)
      af[fm] = ldsA_frag(&sA[wm * 32 + fm * 16 + ln][0], g);
#pragma unroll
    for (int fn = 0; fn < 4; ++fn)
      bf_[fn] = ldsB_frag(&sB[wn * 64 + fn * 16 + ln][0], g);
#pragma unroll
    for (int fm = 0; fm < 2; ++fm)
#pragma unroll
      for (int fn = 0; fn < 4; ++fn)
        acc[fm][fn] = wmma_bf16(af[fm], bf_[fn], acc[fm][fn]);
  }

#pragma unroll
  for (int fm = 0; fm < 2; ++fm)
#pragma unroll
    for (int fn = 0; fn < 4; ++fn)
#pragma unroll
      for (int i = 0; i < 8; ++i) {
        int gr = m0 + wm * 32 + fm * 16 + 8 * g + i;   // token 0..8191
        int gc = n0 + wn * 64 + fn * 16 + ln;          // col 0..3071
        float v = acc[fm][fn][i] + bias[gc];
        int which = gc >> 10, rem = gc & 1023;
        int h = rem >> 6, dh = rem & 63;
        int bt = gr >> 10, nt = gr & 1023;
        size_t plane = ((size_t)which * 128 + bt * 16 + h) << 16;
        size_t idx = plane + (which == 2 ? (((size_t)dh << 10) + nt)
                                         : (((size_t)nt << 6) + dh));
        qkv[idx] = f2bf(v);
      }
}

// ---------------------------------------------------------------------------
// 4) Flash attention per (b,h): grid (16 qblocks, 128 bh), 4 waves x 16 queries
//    Q,K planes [1024][64]; V plane transposed [64][1024].
// ---------------------------------------------------------------------------
__global__ __launch_bounds__(128) void attn_kernel(const __bf16* __restrict__ qkv,
                                                   __bf16* __restrict__ msa) {
  __shared__ __bf16 sQ[64][64];     // [query][dh]
  __shared__ __bf16 sK[64][64];     // [key][dh]
  __shared__ __bf16 sVt[64][64];    // [dh][key]  (global V already transposed)
  __shared__ __bf16 sP[4][16][64];  // per-wave P tile [query][key]
  const int t = threadIdx.x;
  const int lane = t & 31, wave = t >> 5;
  const int g = lane >> 4, ln = lane & 15;
  const int q0 = blockIdx.x * 64;
  const int bh = blockIdx.y;                   // b*16 + h
  const __bf16* Q  = qkv + ((size_t)bh << 16);
  const __bf16* K  = qkv + ((size_t)(128 + bh) << 16);
  const __bf16* Vt = qkv + ((size_t)(256 + bh) << 16);

#pragma unroll
  for (int it = 0; it < 4; ++it) {             // stage Q (64x64 bf16)
    int c = t + it * 128;
    int row = c >> 3, col = (c & 7) * 8;
    cp16(&sQ[row][col], Q + (size_t)(q0 + row) * 64 + col);
  }
  cp16_wait();
  __syncthreads();

  v16bf qf[2];
#pragma unroll
  for (int kc = 0; kc < 2; ++kc)
    qf[kc] = ldsA_frag(&sQ[wave * 16 + ln][kc * 32], g);

  float row_max[8], row_sum[8];
  v8f o[4];
#pragma unroll
  for (int r = 0; r < 8; ++r) { row_max[r] = -1e30f; row_sum[r] = 0.0f; }
#pragma unroll
  for (int nf = 0; nf < 4; ++nf) o[nf] = v8f_zero();

  const float scale = 0.125f;                  // 1/sqrt(64)

  for (int kb = 0; kb < 1024; kb += 64) {
    __syncthreads();
#pragma unroll
    for (int it = 0; it < 4; ++it) {           // stage K and Vt tiles
      int c = t + it * 128;
      int row = c >> 3, col = (c & 7) * 8;
      cp16(&sK[row][col],  K  + (size_t)(kb + row) * 64 + col);
      cp16(&sVt[row][col], Vt + (size_t)row * 1024 + kb + col);
    }
    cp16_wait();
    __syncthreads();

    // S = Q K^T : 16 queries x 64 keys per wave (K rows are B columns over dh)
    v8f s[4];
#pragma unroll
    for (int nf = 0; nf < 4; ++nf) s[nf] = v8f_zero();
#pragma unroll
    for (int kc = 0; kc < 2; ++kc)
#pragma unroll
      for (int nf = 0; nf < 4; ++nf) {
        v16bf bf_ = ldsB_frag(&sK[nf * 16 + ln][kc * 32], g);
        s[nf] = wmma_bf16(qf[kc], bf_, s[nf]);
      }

    // online softmax: lane's rows are (r + 8*laneGroup); columns span the
    // 16-lane half-group, so xor-shuffles 1..8 reduce within each row.
#pragma unroll
    for (int r = 0; r < 8; ++r) {
      float mloc = -1e30f;
#pragma unroll
      for (int nf = 0; nf < 4; ++nf) mloc = fmaxf(mloc, s[nf][r] * scale);
#pragma unroll
      for (int off = 1; off < 16; off <<= 1)
        mloc = fmaxf(mloc, __shfl_xor(mloc, off, 32));
      float mnew  = fmaxf(row_max[r], mloc);
      float alpha = __expf(row_max[r] - mnew);
      float psum = 0.0f;
#pragma unroll
      for (int nf = 0; nf < 4; ++nf) {
        float p = __expf(s[nf][r] * scale - mnew);
        psum += p;
        sP[wave][r + 8 * g][nf * 16 + ln] = f2bf(p);
      }
#pragma unroll
      for (int off = 1; off < 16; off <<= 1)
        psum += __shfl_xor(psum, off, 32);
      row_sum[r] = row_sum[r] * alpha + psum;
      row_max[r] = mnew;
#pragma unroll
      for (int nf = 0; nf < 4; ++nf) o[nf][r] *= alpha;
    }

    // O += P V  (same-wave LDS round trip of P; in-order per wave)
#pragma unroll
    for (int kc = 0; kc < 2; ++kc) {
      v16bf af = ldsA_frag(&sP[wave][ln][kc * 32], g);
#pragma unroll
      for (int nf = 0; nf < 4; ++nf) {
        v16bf bf_ = ldsB_frag(&sVt[nf * 16 + ln][kc * 32], g);
        o[nf] = wmma_bf16(af, bf_, o[nf]);
      }
    }
  }

  const int b = bh >> 4, h = bh & 15;
#pragma unroll
  for (int nf = 0; nf < 4; ++nf)
#pragma unroll
    for (int i = 0; i < 8; ++i) {
      int qr  = q0 + wave * 16 + i + 8 * g;
      int col = h * 64 + nf * 16 + ln;
      float val = o[nf][i] / row_sum[i];
      msa[((size_t)b * 1024 + qr) * 1024 + col] = f2bf(val);
    }
}

// ---------------------------------------------------------------------------
// 5) Proj GEMM + bias + residual: out = msa x Wproj^T + b_proj + z  (f32 out)
// ---------------------------------------------------------------------------
__global__ __launch_bounds__(256) void proj_gemm(const __bf16* __restrict__ A,
                                                 const __bf16* __restrict__ Bt,
                                                 const float* __restrict__ bias,
                                                 const float* __restrict__ z,
                                                 float* __restrict__ out) {
  __shared__ __bf16 sA[128][32];
  __shared__ __bf16 sB[128][32];
  const int t = threadIdx.x;
  const int lane = t & 31, wave = t >> 5;
  const int g = lane >> 4, ln = lane & 15;
  const int wm = wave >> 1, wn = wave & 1;
  const int m0 = blockIdx.x * 128, n0 = blockIdx.y * 128;

  v8f acc[2][4];
#pragma unroll
  for (int i = 0; i < 2; ++i)
#pragma unroll
    for (int j = 0; j < 4; ++j) acc[i][j] = v8f_zero();

  for (int kt = 0; kt < 1024; kt += 32) {
    __syncthreads();
#pragma unroll
    for (int it = 0; it < 2; ++it) {
      int c = t + it * 256;
      int row = c >> 2, col = (c & 3) * 8;
      cp16(&sA[row][col], A  + (size_t)(m0 + row) * 1024 + kt + col);
      cp16(&sB[row][col], Bt + (size_t)(n0 + row) * 1024 + kt + col);
    }
    if (kt + 32 < 1024) {
      __builtin_prefetch(A  + (size_t)(m0 + (t >> 1)) * 1024 + kt + 32, 0, 3);
      __builtin_prefetch(Bt + (size_t)(n0 + (t >> 1)) * 1024 + kt + 32, 0, 3);
    }
    cp16_wait();
    __syncthreads();

    v16bf af[2], bf_[4];
#pragma unroll
    for (int fm = 0; fm < 2; ++fm)
      af[fm] = ldsA_frag(&sA[wm * 32 + fm * 16 + ln][0], g);
#pragma unroll
    for (int fn = 0; fn < 4; ++fn)
      bf_[fn] = ldsB_frag(&sB[wn * 64 + fn * 16 + ln][0], g);
#pragma unroll
    for (int fm = 0; fm < 2; ++fm)
#pragma unroll
      for (int fn = 0; fn < 4; ++fn)
        acc[fm][fn] = wmma_bf16(af[fm], bf_[fn], acc[fm][fn]);
  }

#pragma unroll
  for (int fm = 0; fm < 2; ++fm)
#pragma unroll
    for (int fn = 0; fn < 4; ++fn)
#pragma unroll
      for (int i = 0; i < 8; ++i) {
        int gr = m0 + wm * 32 + fm * 16 + 8 * g + i;
        int gc = n0 + wn * 64 + fn * 16 + ln;
        size_t idx = (size_t)gr * 1024 + gc;
        out[idx] = acc[fm][fn][i] + bias[gc] + z[idx];
      }
}

// ---------------------------------------------------------------------------
// Launch
// ---------------------------------------------------------------------------
extern "C" void kernel_launch(void* const* d_in, const int* in_sizes, int n_in,
                              void* d_out, int out_size, void* d_ws, size_t ws_size,
                              hipStream_t stream) {
  const float* z      = (const float*)d_in[0];
  const float* ln_w   = (const float*)d_in[1];
  const float* ln_b   = (const float*)d_in[2];
  const float* W_qkv  = (const float*)d_in[3];
  const float* b_qkv  = (const float*)d_in[4];
  const float* W_proj = (const float*)d_in[5];
  const float* b_proj = (const float*)d_in[6];
  float* out = (float*)d_out;

  // workspace layout (bytes):
  //   zn      bf16 [8192,1024]           @ 0         (16 MB)
  //   wqkv_t  bf16 [3072,1024]           @ 16777216  ( 6 MB)
  //   wproj_t bf16 [1024,1024]           @ 23068672  ( 2 MB)
  //   qkv     bf16 [3][128] planes 64K   @ 25165824  (48 MB)  (v planes [dh][n])
  //   msa     bf16 [8192,1024]           @ 75497472  (16 MB)
  char* ws = (char*)d_ws;
  __bf16* zn    = (__bf16*)(ws + 0);
  __bf16* wqkvT = (__bf16*)(ws + 16777216);
  __bf16* wprojT= (__bf16*)(ws + 23068672);
  __bf16* qkv   = (__bf16*)(ws + 25165824);
  __bf16* msa   = (__bf16*)(ws + 75497472);

  cvt_t<<<dim3(96, 32), 256, 0, stream>>>(W_qkv, wqkvT, 1024, 3072);
  cvt_t<<<dim3(32, 32), 256, 0, stream>>>(W_proj, wprojT, 1024, 1024);
  ln_kernel<<<8192, 256, 0, stream>>>(z, ln_w, ln_b, zn);
  qkv_gemm<<<dim3(64, 24), 256, 0, stream>>>(zn, wqkvT, b_qkv, qkv);
  attn_kernel<<<dim3(16, 128), 128, 0, stream>>>(qkv, msa);
  proj_gemm<<<dim3(64, 8), 256, 0, stream>>>(msa, wprojT, b_proj, z, out);
}
